// GraphGNN_73658689126809
// MI455X (gfx1250) — matrix operand order repaired and checked by reading, not compiled, not run on hardware
//
#include <hip/hip_runtime.h>
#include <hip/hip_bf16.h>

// GraphConv GNN for MI455X (gfx1250, wave32).
// - Dense GEMMs: v_wmma_f32_16x16x32_f16 (fp32 accumulate).
// - Node features stored f16 (matches WMMA input precision): halves edge gather bytes.
// - Edge scatter: global_atomic_pk_add_f16 via inline asm (2 halves/atomic op,
//   16 ops/edge instead of 32 fp32 ops). h/agg ~6.4MB each -> fully L2-resident.

typedef __attribute__((ext_vector_type(16))) _Float16 v16h;
typedef __attribute__((ext_vector_type(8)))  _Float16 v8h;
typedef __attribute__((ext_vector_type(8)))  float    v8f;

#define HID 32
#define FEAT 128
#define NCLS 16

static __device__ inline v8f wmma_f16(v16h a, v16h b, v8f c) {
  // 8 args: (neg_a, A, neg_b, B, c_mod, C, reuse_a, reuse_b)
  return __builtin_amdgcn_wmma_f32_16x16x32_f16(false, a, false, b, (short)0, c,
                                                false, false);
}

// A tile (16x32 f16) from an f16 row of 32 halves (64B, one half-cacheline).
// A layout used here (contiguous k, same as loadB_f): lanes 0-15: M=lane, halves[0..15]=K 0..15;
// lanes 16-31: same M, K 16..31.  Two 16B-aligned b128 loads per lane, no cvt.
static __device__ inline v16h loadA_h(const _Float16* __restrict__ row, int lh) {
  const v8h lo = *(const v8h*)(row + 16 * lh);
  const v8h hi = *(const v8h*)(row + 16 * lh + 8);
  v16h a;
#pragma unroll
  for (int i = 0; i < 8; ++i) { a[i] = lo[i]; a[8 + i] = hi[i]; }
  return a;
}

// A tile from fp32 row (used only for the x @ W_lin^T input GEMM).
static __device__ inline v16h loadA_f(const float* __restrict__ row, int lh, int k0) {
  v16h a;
#pragma unroll
  for (int i = 0; i < 8; ++i) {
    a[i]     = (_Float16)row[k0 + 16 * lh + i];
    a[8 + i] = (_Float16)row[k0 + 16 * lh + 8 + i];
  }
  return a;
}

// B tile (32x16 f16), B[k][n] = W[n][k]; per lane n=lane&15, halves i hold
// K = 16*(lane>>4)+i -> 16 contiguous floats of W's row n (weights are tiny/cached).
static __device__ inline v16h loadB_f(const float* __restrict__ wrow16) {
  v16h b;
#pragma unroll
  for (int i = 0; i < 16; ++i) b[i] = (_Float16)wrow16[i];
  return b;
}

// ---------------- Stage 1: h0 = relu(x @ W_lin^T + b_lin), stored f16 ----------------
__global__ __launch_bounds__(256) void lin_kernel(const float* __restrict__ x,
                                                  const float* __restrict__ W,
                                                  const float* __restrict__ b,
                                                  _Float16* __restrict__ out,
                                                  int nNodes, int nTiles) {
  const int wave = threadIdx.x >> 5, lane = threadIdx.x & 31;
  const int tile = blockIdx.x * 8 + wave;
  if (tile >= nTiles) return;                       // wave-uniform: EXEC stays full
  const int lh = lane >> 4, ln = lane & 15;
  int arow = tile * 16 + ln; if (arow >= nNodes) arow = nNodes - 1;
  const float* __restrict__ ar = x + (size_t)arow * FEAT;
  const float* __restrict__ w0 = W + (size_t)ln * FEAT + 16 * lh;
  const float* __restrict__ w1 = W + (size_t)(16 + ln) * FEAT + 16 * lh;

  v8f acc0 = {}, acc1 = {};
#pragma unroll
  for (int k0 = 0; k0 < FEAT; k0 += 32) {
    v16h a  = loadA_f(ar, lh, k0);
    v16h b0 = loadB_f(w0 + k0);
    v16h b1 = loadB_f(w1 + k0);
    acc0 = wmma_f16(a, b0, acc0);
    acc1 = wmma_f16(a, b1, acc1);
  }
  const float bias0 = b[ln], bias1 = b[16 + ln];
#pragma unroll
  for (int r = 0; r < 8; ++r) {
    int gm = tile * 16 + r + 8 * lh;
    if (gm < nNodes) {
      _Float16* o = out + (size_t)gm * HID;
      o[ln]      = (_Float16)fmaxf(acc0[r] + bias0, 0.f);
      o[16 + ln] = (_Float16)fmaxf(acc1[r] + bias1, 0.f);
    }
  }
}

// ---------------- Stage 2: agg[dst] += h[src] (f16, packed atomics) ----------------
__global__ __launch_bounds__(256) void scatter_kernel(const _Float16* __restrict__ h,
                                                      const int* __restrict__ edge,
                                                      int nEdges,
                                                      _Float16* __restrict__ agg) {
  int e = blockIdx.x * blockDim.x + threadIdx.x;
  if (e >= nEdges) return;
  const int src = edge[e];
  const int dst = edge[nEdges + e];
  const unsigned int* __restrict__ hr = (const unsigned int*)(h + (size_t)src * HID);
  unsigned int* __restrict__ ar = (unsigned int*)(agg + (size_t)dst * HID);
#pragma unroll
  for (int q = 0; q < 16; ++q) {               // 16 words = 32 halves = one node row
    unsigned int d = hr[q];                    // 2 packed halves
    // CDNA5 packed f16 atomic add (no return -> STOREcnt), 2 elements/op
    asm volatile("global_atomic_pk_add_f16 %0, %1, off"
                 :: "v"(ar + q), "v"(d) : "memory");
  }
}

// --- Stage 3: out = relu(agg@Wrel^T + brel + h@Wroot^T) + fw*first (f16 in/out) ---
__global__ __launch_bounds__(256) void conv_kernel(const _Float16* __restrict__ agg,
                                                   const _Float16* __restrict__ h,
                                                   const float* __restrict__ Wrel,
                                                   const float* __restrict__ brel,
                                                   const float* __restrict__ Wroot,
                                                   const _Float16* __restrict__ first,
                                                   const float* __restrict__ fuse_w,
                                                   int fuse_idx,
                                                   _Float16* __restrict__ out,
                                                   int nNodes, int nTiles) {
  const int wave = threadIdx.x >> 5, lane = threadIdx.x & 31;
  const int tile = blockIdx.x * 8 + wave;
  if (tile >= nTiles) return;
  const int lh = lane >> 4, ln = lane & 15;
  int arow = tile * 16 + ln; if (arow >= nNodes) arow = nNodes - 1;

  v16h a_agg = loadA_h(agg + (size_t)arow * HID, lh);
  v16h a_h   = loadA_h(h   + (size_t)arow * HID, lh);
  v16h brl0  = loadB_f(Wrel  + (size_t)ln        * HID + 16 * lh);
  v16h brl1  = loadB_f(Wrel  + (size_t)(16 + ln) * HID + 16 * lh);
  v16h brt0  = loadB_f(Wroot + (size_t)ln        * HID + 16 * lh);
  v16h brt1  = loadB_f(Wroot + (size_t)(16 + ln) * HID + 16 * lh);

  v8f acc0 = {}, acc1 = {};
  acc0 = wmma_f16(a_agg, brl0, acc0);
  acc0 = wmma_f16(a_h,   brt0, acc0);
  acc1 = wmma_f16(a_agg, brl1, acc1);
  acc1 = wmma_f16(a_h,   brt1, acc1);

  const float fw = fuse_w[fuse_idx];
  const float bias0 = brel[ln], bias1 = brel[16 + ln];
#pragma unroll
  for (int r = 0; r < 8; ++r) {
    int gm = tile * 16 + r + 8 * lh;
    if (gm < nNodes) {
      const _Float16* f = first + (size_t)gm * HID;
      _Float16* o = out + (size_t)gm * HID;
      o[ln]      = (_Float16)(fmaxf(acc0[r] + bias0, 0.f) + fw * (float)f[ln]);
      o[16 + ln] = (_Float16)(fmaxf(acc1[r] + bias1, 0.f) + fw * (float)f[16 + ln]);
    }
  }
}

// ---------------- Stage 4: logits = h @ W_out^T + b_out (fp32 out) ----------------
__global__ __launch_bounds__(256) void out_kernel(const _Float16* __restrict__ h,
                                                  const float* __restrict__ W,
                                                  const float* __restrict__ b,
                                                  float* __restrict__ logits,
                                                  int nNodes, int nTiles) {
  const int wave = threadIdx.x >> 5, lane = threadIdx.x & 31;
  const int tile = blockIdx.x * 8 + wave;
  if (tile >= nTiles) return;
  const int lh = lane >> 4, ln = lane & 15;
  int arow = tile * 16 + ln; if (arow >= nNodes) arow = nNodes - 1;

  v16h a  = loadA_h(h + (size_t)arow * HID, lh);
  v16h bw = loadB_f(W + (size_t)ln * HID + 16 * lh);   // 16 classes -> one N-tile
  v8f acc = {};
  acc = wmma_f16(a, bw, acc);

  const float bias = b[ln];
#pragma unroll
  for (int r = 0; r < 8; ++r) {
    int gm = tile * 16 + r + 8 * lh;
    if (gm < nNodes) logits[(size_t)gm * NCLS + ln] = acc[r] + bias;
  }
}

// ---------------- Stage 5: log_softmax over 16 classes (fp32) ----------------
__global__ __launch_bounds__(256) void lsm_kernel(const float* __restrict__ logits,
                                                  float* __restrict__ out, int nNodes) {
  int i = blockIdx.x * blockDim.x + threadIdx.x;
  if (i >= nNodes) return;
  const float* r = logits + (size_t)i * NCLS;
  float v[NCLS], m = r[0];
#pragma unroll
  for (int c = 0; c < NCLS; ++c) { v[c] = r[c]; m = fmaxf(m, v[c]); }
  float s = 0.f;
#pragma unroll
  for (int c = 0; c < NCLS; ++c) s += expf(v[c] - m);
  const float ls = logf(s);
  float* o = out + (size_t)i * NCLS;
#pragma unroll
  for (int c = 0; c < NCLS; ++c) o[c] = v[c] - m - ls;
}

extern "C" void kernel_launch(void* const* d_in, const int* in_sizes, int n_in,
                              void* d_out, int out_size, void* d_ws, size_t ws_size,
                              hipStream_t stream) {
  const float* x       = (const float*)d_in[0];
  const int*   edge    = (const int*)d_in[1];     // [2, E] (JAX default: int32)
  const float* W_lin   = (const float*)d_in[2];
  const float* b_lin   = (const float*)d_in[3];
  const float* W1_rel  = (const float*)d_in[4];
  const float* b1_rel  = (const float*)d_in[5];
  const float* W1_root = (const float*)d_in[6];
  const float* W2_rel  = (const float*)d_in[7];
  const float* b2_rel  = (const float*)d_in[8];
  const float* W2_root = (const float*)d_in[9];
  const float* fuse_w  = (const float*)d_in[10];
  const float* W_out   = (const float*)d_in[11];
  const float* b_out   = (const float*)d_in[12];

  const int N = in_sizes[0] / FEAT;      // 100000
  const int E = in_sizes[1] / 2;         // 2500000

  const size_t nodeBytesH = (size_t)N * HID * sizeof(_Float16);   // 6.4 MB
  char* ws = (char*)d_ws;
  _Float16* first = (_Float16*)ws;            ws += nodeBytesH;
  _Float16* h1    = (_Float16*)ws;            ws += nodeBytesH;
  _Float16* h2    = (_Float16*)ws;            ws += nodeBytesH;
  _Float16* agg   = (_Float16*)ws;            ws += nodeBytesH;
  float* logits   = (float*)ws;               // 6.4 MB fp32, after the f16 buffers

  const int nTiles   = (N + 15) / 16;
  const int gemmBlks = (nTiles + 7) / 8;           // 8 waves (m-tiles) per 256-thr block
  const int edgeBlks = (E + 255) / 256;
  const int nodeBlks = (N + 255) / 256;

  // h0 = relu(x @ W_lin^T + b)
  lin_kernel<<<gemmBlks, 256, 0, stream>>>(x, W_lin, b_lin, first, N, nTiles);

  // conv1
  hipMemsetAsync(agg, 0, nodeBytesH, stream);
  scatter_kernel<<<edgeBlks, 256, 0, stream>>>(first, edge, E, agg);
  conv_kernel<<<gemmBlks, 256, 0, stream>>>(agg, first, W1_rel, b1_rel, W1_root,
                                            first, fuse_w, 0, h1, N, nTiles);
  // conv2
  hipMemsetAsync(agg, 0, nodeBytesH, stream);
  scatter_kernel<<<edgeBlks, 256, 0, stream>>>(h1, edge, E, agg);
  conv_kernel<<<gemmBlks, 256, 0, stream>>>(agg, h1, W2_rel, b2_rel, W2_root,
                                            first, fuse_w, 1, h2, N, nTiles);

  // logits + log_softmax
  out_kernel<<<gemmBlks, 256, 0, stream>>>(h2, W_out, b_out, logits, N, nTiles);
  lsm_kernel<<<nodeBlks, 256, 0, stream>>>(logits, (float*)d_out, N);
}
